// SModel_26276609917535
// MI455X (gfx1250) — matrix operand order repaired and checked by reading, not compiled, not run on hardware
//
#include <hip/hip_runtime.h>
#include <hip/hip_bf16.h>

// ---------------------------------------------------------------------------
// Problem constants (from reference)
// ---------------------------------------------------------------------------
#define S_NODES 50000
#define E_EDGES 400000
#define FD 128
#define FM 256      // 2*FD
#define FM2 1280    // 4*FM + 2*FD

typedef __attribute__((ext_vector_type(16))) __bf16 bf16x16;
typedef __attribute__((ext_vector_type(8)))  float  f32x8;

__device__ inline void atomic_add_f32(float* p, float v) {
    __hip_atomic_fetch_add(p, v, __ATOMIC_RELAXED, __HIP_MEMORY_SCOPE_AGENT);
}

// jnp.nan_to_num: NaN->0, +/-inf -> +/-FLT_MAX
__device__ inline float fixnum(float x) {
    if (x != x) return 0.0f;
    const float big = 3.4028234e38f;
    if (x >  big) return  big;
    if (x < -big) return -big;
    return x;
}

// ---------------------------------------------------------------------------
// Helper kernels
// ---------------------------------------------------------------------------
__global__ void zero_f32(float* __restrict__ p, long long n) {
    long long i = (long long)blockIdx.x * blockDim.x + threadIdx.x;
    long long stride = (long long)gridDim.x * blockDim.x;
    for (; i < n; i += stride) p[i] = 0.0f;
}

__global__ void cvt_bf16(const float* __restrict__ in, __bf16* __restrict__ out, int n) {
    int i = blockIdx.x * blockDim.x + threadIdx.x;
    if (i < n) out[i] = (__bf16)in[i];
}

// ---------------------------------------------------------------------------
// CSR construction: degree histogram -> exclusive scan -> edge-id scatter
// ---------------------------------------------------------------------------
__global__ void deg_kernel(const long long* __restrict__ ei, int* __restrict__ deg) {
    int e = blockIdx.x * blockDim.x + threadIdx.x;
    if (e < E_EDGES) atomicAdd(&deg[(int)ei[e]], 1);
}

#define SCAN_T 256
__global__ __launch_bounds__(SCAN_T)
void scan_kernel(const int* __restrict__ deg, int* __restrict__ rowstart,
                 int* __restrict__ cursor) {
    __shared__ int part[SCAN_T];
    __shared__ int base_s[SCAN_T];
    const int tid = threadIdx.x;
    const int CH  = (S_NODES + SCAN_T - 1) / SCAN_T;   // 196 nodes per thread
    const int b0  = tid * CH;
    int sum = 0;
    for (int i = 0; i < CH; ++i) {
        int g = b0 + i;
        if (g < S_NODES) sum += deg[g];
    }
    part[tid] = sum;
    __syncthreads();
    if (tid == 0) {
        int run = 0;
        for (int i = 0; i < SCAN_T; ++i) { int t = part[i]; base_s[i] = run; run += t; }
        rowstart[S_NODES] = run;
    }
    __syncthreads();
    int run = base_s[tid];
    for (int i = 0; i < CH; ++i) {
        int g = b0 + i;
        if (g < S_NODES) { rowstart[g] = run; cursor[g] = run; run += deg[g]; }
    }
}

__global__ void scatter_kernel(const long long* __restrict__ ei,
                               int* __restrict__ cursor, int* __restrict__ csr) {
    int e = blockIdx.x * blockDim.x + threadIdx.x;
    if (e < E_EDGES) {
        int s = (int)ei[e];
        int pos = atomicAdd(&cursor[s], 1);
        csr[pos] = e;
    }
}

// ---------------------------------------------------------------------------
// Edge kernel: msg = MLP1(concat(x_t[tgt], edge_attr)) -> stream msg to global.
// Block = 128 threads (4 waves), one 32-edge tile per block.
// Wave owns 4 N-tiles x 2 M-tiles (4 waves * 4 * 16 = 256 cols, 2*16 = 32 rows)
// => each B fragment is reused for 2 WMMAs (halves L2 weight traffic).
// ---------------------------------------------------------------------------
__global__ __launch_bounds__(128)
void edge_kernel(const float* __restrict__ x_t, const float* __restrict__ edge_attr,
                 const long long* __restrict__ ei,           // [2, E]
                 const __bf16* __restrict__ w1a, const float* __restrict__ b1a,
                 const __bf16* __restrict__ w1b, const float* __restrict__ b1b,
                 float* __restrict__ msg)
{
    __shared__ __bf16 Ain[32][FM];   // 16 KB input tile (bf16)
    __shared__ __bf16 H1 [32][FM];   // 16 KB hidden tile (bf16)

    const int tid  = threadIdx.x;
    const int lane = tid & 31;
    const int wave = tid >> 5;
    const int half = lane >> 4;      // 0: lanes 0-15, 1: lanes 16-31
    const int lcol = lane & 15;
    const long long ebase = (long long)blockIdx.x * 32;

    // ---- stage input tile: concat(x_t[tgt], edge_attr) as bf16 -------------
    {
        int row = tid >> 2;              // 32 rows, 4 threads/row
        int cb  = (tid & 3) * 64;        // 64 consecutive features each
        long long e = ebase + row;
        long long tgt = ei[E_EDGES + e];
        const float* xr = x_t + tgt * FD;
        const float* ar = edge_attr + e * FD;
        #pragma unroll
        for (int i = 0; i < 64; ++i) {
            int c = cb + i;
            float v = (c < FD) ? xr[c] : ar[c - FD];
            Ain[row][c] = (__bf16)v;
        }
    }
    __syncthreads();

    // ---- GEMM1: [32x256] x W1a^T -> leaky(0.1) -> H1 -----------------------
    f32x8 acc[2][4];
    #pragma unroll
    for (int t = 0; t < 2; ++t)
        #pragma unroll
        for (int j = 0; j < 4; ++j)
            #pragma unroll
            for (int v = 0; v < 8; ++v) acc[t][j][v] = 0.0f;

    for (int kk = 0; kk < FM / 32; ++kk) {
        const int k0 = kk * 32;
        bf16x16 a[2];
        #pragma unroll
        for (int t = 0; t < 2; ++t) {
            const __bf16* p0 = &Ain[t * 16 + lcol][k0 + half * 16];
            #pragma unroll
            for (int i = 0; i < 16; ++i) a[t][i] = p0[i];
        }
        #pragma unroll
        for (int j = 0; j < 4; ++j) {
            const int n = (wave * 4 + j) * 16 + lcol;     // B col = W row
            const __bf16* p = w1a + n * FM + k0 + half * 16;
            bf16x16 b;
            #pragma unroll
            for (int i = 0; i < 16; ++i) b[i] = p[i];
            #pragma unroll
            for (int t = 0; t < 2; ++t)
                acc[t][j] = __builtin_amdgcn_wmma_f32_16x16x32_bf16(
                                false, a[t], false, b, (short)0, acc[t][j], false, false);
        }
    }
    #pragma unroll
    for (int t = 0; t < 2; ++t)
        #pragma unroll
        for (int j = 0; j < 4; ++j) {
            const int n = (wave * 4 + j) * 16 + lcol;
            const float bias = b1a[n];
            #pragma unroll
            for (int v = 0; v < 8; ++v) {
                float x = acc[t][j][v] + bias;
                x = (x < 0.0f) ? 0.1f * x : x;
                H1[t * 16 + v + 8 * half][n] = (__bf16)x;  // C layout: M = vgpr + 8*half
            }
        }
    __syncthreads();

    // ---- GEMM2: H1 x W1b^T + b1b -> stream msg (f32) -----------------------
    f32x8 acc2[2][4];
    #pragma unroll
    for (int t = 0; t < 2; ++t)
        #pragma unroll
        for (int j = 0; j < 4; ++j)
            #pragma unroll
            for (int v = 0; v < 8; ++v) acc2[t][j][v] = 0.0f;

    for (int kk = 0; kk < FM / 32; ++kk) {
        const int k0 = kk * 32;
        bf16x16 a[2];
        #pragma unroll
        for (int t = 0; t < 2; ++t) {
            const __bf16* p0 = &H1[t * 16 + lcol][k0 + half * 16];
            #pragma unroll
            for (int i = 0; i < 16; ++i) a[t][i] = p0[i];
        }
        #pragma unroll
        for (int j = 0; j < 4; ++j) {
            const int n = (wave * 4 + j) * 16 + lcol;
            const __bf16* p = w1b + n * FM + k0 + half * 16;
            bf16x16 b;
            #pragma unroll
            for (int i = 0; i < 16; ++i) b[i] = p[i];
            #pragma unroll
            for (int t = 0; t < 2; ++t)
                acc2[t][j] = __builtin_amdgcn_wmma_f32_16x16x32_bf16(
                                 false, a[t], false, b, (short)0, acc2[t][j], false, false);
        }
    }
    #pragma unroll
    for (int t = 0; t < 2; ++t)
        #pragma unroll
        for (int j = 0; j < 4; ++j) {
            const int n = (wave * 4 + j) * 16 + lcol;
            const float bias = b1b[n];
            #pragma unroll
            for (int v = 0; v < 8; ++v) {
                const int m = t * 16 + v + 8 * half;
                msg[(ebase + m) * FM + n] = acc2[t][j][v] + bias;
            }
        }
}

// ---------------------------------------------------------------------------
// Node kernel: gather per-node raw moments of msg through CSR in the staging
// epilogue (mean/std/skew/kurt via central-moment algebra), build the
// [16 x 1280] bf16 feature tile in LDS, then MLP2 via WMMA; BN accumulation.
// Block = 256 threads (8 waves), 16 nodes per block (S % 16 == 0).
// ---------------------------------------------------------------------------
__global__ __launch_bounds__(256)
void node_kernel(const float* __restrict__ x_s, const float* __restrict__ u,
                 const __bf16* __restrict__ w2a, const float* __restrict__ b2a,
                 const __bf16* __restrict__ w2b, const float* __restrict__ b2b,
                 const float* __restrict__ msg,
                 const int* __restrict__ rowstart, const int* __restrict__ csr,
                 float* __restrict__ hbuf, float* __restrict__ bn_sum,
                 float* __restrict__ bn_sqs)
{
    __shared__ __bf16 Abuf[16][FM2];   // 40 KB, reused for hidden activations

    const int tid  = threadIdx.x;
    const int lane = tid & 31;
    const int wave = tid >> 5;
    const int half = lane >> 4;
    const int lcol = lane & 15;
    const long long gbase = (long long)blockIdx.x * 16;

    // ---- stage x_s and broadcast u columns ---------------------------------
    for (int idx = tid; idx < 16 * FD; idx += 256) {
        const int row = idx >> 7;          // /128
        const int c   = idx & 127;
        const long long g = gbase + row;
        Abuf[row][c]              = (__bf16)x_s[g * FD + c];
        Abuf[row][FD + 4 * FM + c] = (__bf16)u[c];
    }
    // ---- stage statistics columns: one CSR gather per (row, feature) -------
    for (int idx = tid; idx < 16 * FM; idx += 256) {
        const int row = idx >> 8;          // /256
        const int j   = idx & 255;
        const long long g = gbase + row;
        const int rs = rowstart[g];
        const int re = rowstart[g + 1];
        float a1 = 0.0f, a2 = 0.0f, a3 = 0.0f, a4 = 0.0f;
        for (int k = rs; k < re; ++k) {
            const int eid = csr[k];
            const float v  = msg[(long long)eid * FM + j];
            const float v2 = v * v;
            a1 += v; a2 += v2; a3 += v2 * v; a4 += v2 * v2;
        }
        const float dn = fmaxf((float)(re - rs), 1.0f);
        const float m1 = a1 / dn, m2 = a2 / dn, m3 = a3 / dn, m4 = a4 / dn;
        const float mean = fixnum(m1);
        float var = m2 - m1 * m1;
        var = (var < 0.0f) ? 0.01f * var : var;       // leaky(0.01)
        var = fixnum(var);
        const float sp   = var + 1e-6f;               // std^2
        const float stdv = sqrtf(sp);
        const float m1sq = m1 * m1;
        const float m3c  = m3 - 3.0f * m1 * m2 + 2.0f * m1 * m1sq;
        const float m4c  = m4 - 4.0f * m1 * m3 + 6.0f * m1sq * m2 - 3.0f * m1sq * m1sq;
        const float skew = fixnum(m3c / (sp * stdv)); // / std^3
        const float kurt = fixnum(m4c / (sp * sp));   // / std^4
        Abuf[row][FD + j]          = (__bf16)mean;
        Abuf[row][FD + FM + j]     = (__bf16)stdv;
        Abuf[row][FD + 2 * FM + j] = (__bf16)skew;
        Abuf[row][FD + 3 * FM + j] = (__bf16)kurt;
    }
    __syncthreads();

    // ---- GEMM A: [16x1280] x W2a^T, accumulators stay in VGPRs -------------
    f32x8 acc[10];
    #pragma unroll
    for (int j = 0; j < 10; ++j)
        #pragma unroll
        for (int v = 0; v < 8; ++v) acc[j][v] = 0.0f;

    for (int kk = 0; kk < FM2 / 32; ++kk) {
        const int k0 = kk * 32;
        bf16x16 a;
        {
            const __bf16* p0 = &Abuf[lcol][k0 + half * 16];
            #pragma unroll
            for (int i = 0; i < 16; ++i) a[i] = p0[i];
        }
        #pragma unroll
        for (int j = 0; j < 10; ++j) {
            const int n = (wave * 10 + j) * 16 + lcol;
            const __bf16* p = w2a + (long long)n * FM2 + k0 + half * 16;
            bf16x16 b;
            #pragma unroll
            for (int i = 0; i < 16; ++i) b[i] = p[i];
            acc[j] = __builtin_amdgcn_wmma_f32_16x16x32_bf16(
                         false, a, false, b, (short)0, acc[j], false, false);
        }
    }
    __syncthreads();   // everyone done READING Abuf
    #pragma unroll
    for (int j = 0; j < 10; ++j) {
        const int n = (wave * 10 + j) * 16 + lcol;
        const float bias = b2a[n];
        #pragma unroll
        for (int v = 0; v < 8; ++v) {
            float x = acc[j][v] + bias;
            x = (x < 0.0f) ? 0.1f * x : x;
            Abuf[v + 8 * half][n] = (__bf16)x;   // overwrite with hidden acts
        }
    }
    __syncthreads();

    // ---- GEMM B: [16x1280] x W2b^T -> h[16x128]; BN accumulation -----------
    f32x8 c2;
    #pragma unroll
    for (int v = 0; v < 8; ++v) c2[v] = 0.0f;

    for (int kk = 0; kk < FM2 / 32; ++kk) {
        const int k0 = kk * 32;
        bf16x16 a;
        {
            const __bf16* p0 = &Abuf[lcol][k0 + half * 16];
            #pragma unroll
            for (int i = 0; i < 16; ++i) a[i] = p0[i];
        }
        const int n = wave * 16 + lcol;
        const __bf16* p = w2b + (long long)n * FM2 + k0 + half * 16;
        bf16x16 b;
        #pragma unroll
        for (int i = 0; i < 16; ++i) b[i] = p[i];
        c2 = __builtin_amdgcn_wmma_f32_16x16x32_bf16(
                 false, a, false, b, (short)0, c2, false, false);
    }
    {
        const int n = wave * 16 + lcol;
        const float bias = b2b[n];
        float psum = 0.0f, psq = 0.0f;
        #pragma unroll
        for (int v = 0; v < 8; ++v) {
            const float x = c2[v] + bias;
            const int m = v + 8 * half;
            hbuf[(gbase + m) * FD + n] = x;
            psum += x;
            psq  += x * x;
        }
        atomic_add_f32(&bn_sum[n], psum);
        atomic_add_f32(&bn_sqs[n], psq);
    }
}

// ---------------------------------------------------------------------------
// BatchNorm finalize (training-mode batch stats, biased variance)
// ---------------------------------------------------------------------------
__global__ void bn_kernel(const float* __restrict__ hbuf,
                          const float* __restrict__ bn_sum,
                          const float* __restrict__ bn_sqs,
                          const float* __restrict__ gamma,
                          const float* __restrict__ beta,
                          float* __restrict__ out)
{
    long long i = (long long)blockIdx.x * blockDim.x + threadIdx.x;
    if (i >= (long long)S_NODES * FD) return;
    const int c = (int)(i % FD);
    const float inv_n = 1.0f / (float)S_NODES;
    const float mu  = bn_sum[c] * inv_n;
    const float var = bn_sqs[c] * inv_n - mu * mu;
    out[i] = (hbuf[i] - mu) * rsqrtf(var + 1e-5f) * gamma[c] + beta[c];
}

// ---------------------------------------------------------------------------
// Launch
// ---------------------------------------------------------------------------
extern "C" void kernel_launch(void* const* d_in, const int* in_sizes, int n_in,
                              void* d_out, int out_size, void* d_ws, size_t ws_size,
                              hipStream_t stream) {
    (void)in_sizes; (void)n_in; (void)out_size; (void)ws_size;
    const float*     x_s       = (const float*)d_in[0];
    const float*     x_t       = (const float*)d_in[1];
    const float*     edge_attr = (const float*)d_in[2];
    const float*     u         = (const float*)d_in[3];
    const float*     W1a       = (const float*)d_in[4];
    const float*     b1a       = (const float*)d_in[5];
    const float*     W1b       = (const float*)d_in[6];
    const float*     b1b       = (const float*)d_in[7];
    const float*     W2a       = (const float*)d_in[8];
    const float*     b2a       = (const float*)d_in[9];
    const float*     W2b       = (const float*)d_in[10];
    const float*     b2b       = (const float*)d_in[11];
    const float*     gamma     = (const float*)d_in[12];
    const float*     beta      = (const float*)d_in[13];
    const long long* ei        = (const long long*)d_in[14];
    float*           out       = (float*)d_out;

    // workspace carve-up (256B aligned)
    char* ws = (char*)d_ws;
    size_t off = 0;
    auto take = [&](size_t bytes) -> char* {
        char* p = ws + off;
        off = (off + bytes + 255) & ~(size_t)255;
        return p;
    };
    __bf16* w1a_bf   = (__bf16*)take((size_t)FM  * FM  * 2);
    __bf16* w1b_bf   = (__bf16*)take((size_t)FM  * FM  * 2);
    __bf16* w2a_bf   = (__bf16*)take((size_t)FM2 * FM2 * 2);
    __bf16* w2b_bf   = (__bf16*)take((size_t)FD  * FM2 * 2);
    float*  msg      = (float*) take((size_t)E_EDGES * FM * 4);      // 409.6 MB
    int*    deg      = (int*)   take((size_t)S_NODES * 4);
    int*    rowstart = (int*)   take((size_t)(S_NODES + 1) * 4);
    int*    cursor   = (int*)   take((size_t)S_NODES * 4);
    int*    csr      = (int*)   take((size_t)E_EDGES * 4);
    float*  hbuf     = (float*) take((size_t)S_NODES * FD * 4);
    float*  bn_sum   = (float*) take((size_t)FD * 4);
    float*  bn_sqs   = (float*) take((size_t)FD * 4);

    // 1) zero the accumulators that are actually accumulated into
    zero_f32<<<(S_NODES + 255) / 256, 256, 0, stream>>>((float*)deg, S_NODES);
    zero_f32<<<1, 256, 0, stream>>>(bn_sum, 2 * FD);   // bn_sum+bn_sqs contiguous
    // 2) weights -> bf16 (all resident in L2 thereafter)
    cvt_bf16<<<(FM * FM + 255) / 256, 256, 0, stream>>>(W1a, w1a_bf, FM * FM);
    cvt_bf16<<<(FM * FM + 255) / 256, 256, 0, stream>>>(W1b, w1b_bf, FM * FM);
    cvt_bf16<<<(FM2 * FM2 + 255) / 256, 256, 0, stream>>>(W2a, w2a_bf, FM2 * FM2);
    cvt_bf16<<<(FD * FM2 + 255) / 256, 256, 0, stream>>>(W2b, w2b_bf, FD * FM2);
    // 3) CSR: degree -> scan -> scatter
    deg_kernel<<<(E_EDGES + 255) / 256, 256, 0, stream>>>(ei, deg);
    scan_kernel<<<1, SCAN_T, 0, stream>>>(deg, rowstart, cursor);
    scatter_kernel<<<(E_EDGES + 255) / 256, 256, 0, stream>>>(ei, cursor, csr);
    // 4) edge MLP -> stream msg (E/32 = 12500 tiles)
    edge_kernel<<<E_EDGES / 32, 128, 0, stream>>>(x_t, edge_attr, ei,
                                                  w1a_bf, b1a, w1b_bf, b1b, msg);
    // 5) node MLP with fused CSR moment gather (S/16 = 3125 tiles)
    node_kernel<<<S_NODES / 16, 256, 0, stream>>>(x_s, u, w2a_bf, b2a, w2b_bf, b2b,
                                                  msg, rowstart, csr,
                                                  hbuf, bn_sum, bn_sqs);
    // 6) batchnorm finalize
    bn_kernel<<<(S_NODES * FD + 255) / 256, 256, 0, stream>>>(hbuf, bn_sum, bn_sqs,
                                                              gamma, beta, out);
}
